// RAN_36524401885333
// MI455X (gfx1250) — matrix operand order repaired and checked by reading, not compiled, not run on hardware
//
#include <hip/hip_runtime.h>

// Problem constants (from reference)
#define T_STEPS 1024
#define BATCH   64
#define NINP    2048
#define NHID    2048

// ---- vector types matching probe-verified WMMA builtin signatures ----
typedef __attribute__((ext_vector_type(16))) __bf16 bf16x16;
typedef __attribute__((ext_vector_type(8)))  __bf16 bf16x8;
typedef __attribute__((ext_vector_type(8)))  float  f32x8;
typedef __attribute__((ext_vector_type(4)))  float  f32x4;
typedef __attribute__((ext_vector_type(4)))  unsigned int u32x4;

union ABReg {
    bf16x16        v;      // WMMA operand
    bf16x8         h[2];   // two 128-bit halves for global_load_b128
};

__device__ __forceinline__ unsigned short f2bf(float f) {
    union { float f; unsigned u; } v; v.f = f;
    unsigned r = v.u + 0x7FFFu + ((v.u >> 16) & 1u);   // round-to-nearest-even
    return (unsigned short)(r >> 16);
}

// convert 8 consecutive f32 -> 8 bf16, 128-bit in / 128-bit out
__device__ __forceinline__ void cvt_store8(const float* __restrict__ src,
                                           unsigned short* __restrict__ dst) {
    f32x4 a = *(const f32x4*)(src);
    f32x4 b = *(const f32x4*)(src + 4);
    u32x4 p;
    p.x = (unsigned)f2bf(a.x) | ((unsigned)f2bf(a.y) << 16);
    p.y = (unsigned)f2bf(a.z) | ((unsigned)f2bf(a.w) << 16);
    p.z = (unsigned)f2bf(b.x) | ((unsigned)f2bf(b.y) << 16);
    p.w = (unsigned)f2bf(b.z) | ((unsigned)f2bf(b.w) << 16);
    *(u32x4*)dst = p;
}

__device__ __forceinline__ f32x8 wmma_bf16(bf16x16 a, bf16x16 b, f32x8 c) {
    // 8 args: (neg_a, A, neg_b, B, c_mod, C, reuse_a, reuse_b)
    return __builtin_amdgcn_wmma_f32_16x16x32_bf16(false, a, false, b,
                                                   (short)0, c, false, false);
}

__device__ __forceinline__ float fast_sigmoid(float x) {
    return 1.0f / (1.0f + __expf(-x));
}

// ---------------- workspace layout (bytes) ----------------
static constexpr size_t WMAT_ELEMS  = (size_t)NHID * NINP;            // 4Mi elems / matrix
static constexpr size_t X_ELEMS     = (size_t)T_STEPS * BATCH * NINP; // 134Mi
static constexpr size_t WBF_OFF     = 0;                              // 5 bf16 matrices
static constexpr size_t WBF_BYTES   = 5 * WMAT_ELEMS * 2;             // 41,943,040
static constexpr size_t XBF_OFF     = WBF_OFF + WBF_BYTES;            // bf16 copy of x
static constexpr size_t XBF_BYTES   = X_ELEMS * 2;                    // 268,435,456
static constexpr size_t HBF_OFF     = XBF_OFF + XBF_BYTES;            // double-buffered h (bf16)
static constexpr size_t HBF_BYTES   = 2 * (size_t)BATCH * NHID * 2;
static constexpr size_t HF_OFF      = HBF_OFF + HBF_BYTES;            // double-buffered h (f32)
static constexpr size_t HF_BYTES    = 2 * (size_t)BATCH * NHID * 4;
static constexpr size_t CNT_OFF     = HF_OFF + HF_BYTES;              // grid-barrier counter
static constexpr size_t PROJ_ELEMS  = (size_t)T_STEPS * BATCH * NHID;
static constexpr size_t CONTENT_OFF = CNT_OFF + 256;
static constexpr size_t IX_OFF      = CONTENT_OFF + PROJ_ELEMS * 4;
static constexpr size_t FX_OFF      = IX_OFF      + PROJ_ELEMS * 4;

#define SCAN_BLOCKS 32

// ---------------- kernel 0: zero the barrier counter ----------------
__global__ void init_counter_kernel(unsigned int* counter) {
    if (threadIdx.x == 0) *counter = 0u;
}

// ---------------- kernel 1: fp32 -> bf16 conversions (one streaming pass) ----------------
__global__ void convert_kernel(const float* __restrict__ x,
                               const float* __restrict__ Wcx, const float* __restrict__ Wih,
                               const float* __restrict__ Wix, const float* __restrict__ Wfh,
                               const float* __restrict__ Wfx, const float* __restrict__ hidden,
                               unsigned short* __restrict__ xbf,
                               unsigned short* __restrict__ Wbf,
                               unsigned short* __restrict__ hbf0,
                               float* __restrict__ hf0) {
    const size_t tid    = (size_t)blockIdx.x * blockDim.x + threadIdx.x;
    const size_t stride = (size_t)gridDim.x * blockDim.x;

    // x: 134Mi elements, 8 per thread-iteration
    for (size_t i = tid; i < X_ELEMS / 8; i += stride)
        cvt_store8(x + i * 8, xbf + i * 8);

    // weights: 5 x 4Mi elements
    const size_t n8 = WMAT_ELEMS / 8;
    for (size_t i = tid; i < n8; i += stride) {
        cvt_store8(Wcx + i * 8, Wbf + 0 * WMAT_ELEMS + i * 8);
        cvt_store8(Wih + i * 8, Wbf + 1 * WMAT_ELEMS + i * 8);
        cvt_store8(Wix + i * 8, Wbf + 2 * WMAT_ELEMS + i * 8);
        cvt_store8(Wfh + i * 8, Wbf + 3 * WMAT_ELEMS + i * 8);
        cvt_store8(Wfx + i * 8, Wbf + 4 * WMAT_ELEMS + i * 8);
    }

    // initial hidden state: bf16 + f32 copies into buffer 0
    for (size_t i = tid; i < (size_t)BATCH * NHID; i += stride) {
        float h = hidden[i];
        hbf0[i] = f2bf(h);
        hf0[i]  = h;
    }
}

// ---------------- kernel 2: input-side projection GEMMs ----------------
// out[m, n] = sum_k x[m, k] * W[n, k] (+ bias[n])
// grid: (NHID/64, M/128, 3), block 256 = 8 waves.
// Each wave: 32x32 macro-tile = 2 M-tiles x 2 N-tiles, 4 accumulators.
// Per K-step: 8 x global_load_b128 -> 4 x v_wmma (pure bf16, no conversion).
__global__ void __launch_bounds__(256)
proj_gemm_kernel(const unsigned short* __restrict__ xbf,  // [T*B, NINP] bf16
                 const unsigned short* __restrict__ Wbf,  // 5 bf16 matrices, base
                 const float* __restrict__ b_i, const float* __restrict__ b_f,
                 float* __restrict__ content, float* __restrict__ ixb,
                 float* __restrict__ fxb) {
    const int K = NINP;
    const int which = blockIdx.z;   // 0: content(Wcx) 1: ix(Wix,+b_i) 2: fx(Wfx,+b_f)
    const unsigned short* W;
    float* out;
    const float* bias;
    if (which == 0)      { W = Wbf + 0 * WMAT_ELEMS; out = content; bias = nullptr; }
    else if (which == 1) { W = Wbf + 2 * WMAT_ELEMS; out = ixb;     bias = b_i;     }
    else                 { W = Wbf + 4 * WMAT_ELEMS; out = fxb;     bias = b_f;     }

    const int wave = threadIdx.x >> 5;   // 0..7
    const int lane = threadIdx.x & 31;
    const int mw   = wave & 3;           // 4 waves along M
    const int nw   = wave >> 2;          // 2 waves along N
    const size_t m_base = (size_t)blockIdx.y * 128 + (size_t)mw * 32; // 2 M tiles
    const int    n_base = blockIdx.x * 64 + nw * 32;                  // 2 N tiles

    const int lm = lane & 15;
    const int lh = lane >> 4;

    // A operand rows (K layout: contiguous 16 K-elements per lane, lane>=16 -> +16)
    const unsigned short* arow0 = xbf + (m_base + lm) * (size_t)K + lh * 16;
    const unsigned short* arow1 = arow0 + (size_t)16 * K;
    // B operand rows = W rows (contiguous 16 K-elements per lane)
    const unsigned short* brow0 = W + (size_t)(n_base + lm) * K + lh * 16;
    const unsigned short* brow1 = brow0 + (size_t)16 * K;

    f32x8 acc00 = {}, acc01 = {}, acc10 = {}, acc11 = {};
    #pragma unroll 2
    for (int k0 = 0; k0 < K; k0 += 32) {
        ABReg A0, A1, B0, B1;
        A0.h[0] = *(const bf16x8*)(arow0 + k0);
        A0.h[1] = *(const bf16x8*)(arow0 + k0 + 8);
        A1.h[0] = *(const bf16x8*)(arow1 + k0);
        A1.h[1] = *(const bf16x8*)(arow1 + k0 + 8);
        B0.h[0] = *(const bf16x8*)(brow0 + k0);
        B0.h[1] = *(const bf16x8*)(brow0 + k0 + 8);
        B1.h[0] = *(const bf16x8*)(brow1 + k0);
        B1.h[1] = *(const bf16x8*)(brow1 + k0 + 8);
        acc00 = wmma_bf16(A0.v, B0.v, acc00);
        acc01 = wmma_bf16(A0.v, B1.v, acc01);
        acc10 = wmma_bf16(A1.v, B0.v, acc10);
        acc11 = wmma_bf16(A1.v, B1.v, acc11);
    }

    const float bias0 = bias ? bias[n_base + lm]      : 0.0f;
    const float bias1 = bias ? bias[n_base + 16 + lm] : 0.0f;
    const size_t row0 = m_base + (size_t)lh * 8;   // M tile 0
    #pragma unroll
    for (int e = 0; e < 8; ++e) {
        size_t r0 = (row0 + e) * (size_t)NHID;
        size_t r1 = (row0 + 16 + e) * (size_t)NHID;
        out[r0 + n_base + lm]      = acc00[e] + bias0;
        out[r0 + n_base + 16 + lm] = acc01[e] + bias1;
        out[r1 + n_base + lm]      = acc10[e] + bias0;
        out[r1 + n_base + 16 + lm] = acc11[e] + bias1;
    }
}

// ---------------- kernel 3: persistent recurrent scan ----------------
// 32 blocks x 256 threads = 256 waves. Wave owns 1 M-tile x 2 N-tiles for BOTH
// gate GEMMs (4 accumulators, shared A loads), then the fused sigmoid/tanh
// epilogue. One device-wide barrier per step; h double-buffered.
__global__ void __launch_bounds__(256)
scan_kernel(const unsigned short* __restrict__ Wbf,
            const float* __restrict__ content, const float* __restrict__ ixb,
            const float* __restrict__ fxb,
            unsigned short* __restrict__ hbf, float* __restrict__ hf,
            float* __restrict__ output, float* __restrict__ nexth,
            unsigned int* counter) {
    const int K = NHID;
    const unsigned short* Wih = Wbf + 1 * WMAT_ELEMS;
    const unsigned short* Wfh = Wbf + 3 * WMAT_ELEMS;

    const int wave_g = blockIdx.x * 8 + (threadIdx.x >> 5);  // 0..255
    const int lane   = threadIdx.x & 31;
    const int mi     = wave_g & 3;        // 4 M tiles (B=64)
    const int np     = wave_g >> 2;       // 64 N tile-pairs (H=2048)
    const int m_base = mi * 16;
    const int n_base = np * 32;           // two N tiles: n_base, n_base+16
    const int lm     = lane & 15;
    const int lh     = lane >> 4;

    const unsigned short* bi0 = Wih + (size_t)(n_base + lm) * K + lh * 16;
    const unsigned short* bi1 = bi0 + (size_t)16 * K;
    const unsigned short* bf0 = Wfh + (size_t)(n_base + lm) * K + lh * 16;
    const unsigned short* bf1 = bf0 + (size_t)16 * K;
    const size_t a_off = (size_t)(m_base + lm) * K + lh * 16;

    for (int t = 0; t < T_STEPS; ++t) {
        const unsigned short* hread_bf  = hbf + (size_t)(t & 1) * BATCH * NHID;
        const float*          hread_f   = hf  + (size_t)(t & 1) * BATCH * NHID;
        unsigned short*       hwrite_bf = hbf + (size_t)((t + 1) & 1) * BATCH * NHID;
        float*                hwrite_f  = hf  + (size_t)((t + 1) & 1) * BATCH * NHID;

        const unsigned short* arow = hread_bf + a_off;

        f32x8 ai0 = {}, ai1 = {}, af0 = {}, af1 = {};
        #pragma unroll 2
        for (int k0 = 0; k0 < K; k0 += 32) {
            ABReg A, Bi0, Bi1, Bf0, Bf1;
            A.h[0]   = *(const bf16x8*)(arow + k0);
            A.h[1]   = *(const bf16x8*)(arow + k0 + 8);
            Bi0.h[0] = *(const bf16x8*)(bi0 + k0);
            Bi0.h[1] = *(const bf16x8*)(bi0 + k0 + 8);
            Bi1.h[0] = *(const bf16x8*)(bi1 + k0);
            Bi1.h[1] = *(const bf16x8*)(bi1 + k0 + 8);
            Bf0.h[0] = *(const bf16x8*)(bf0 + k0);
            Bf0.h[1] = *(const bf16x8*)(bf0 + k0 + 8);
            Bf1.h[0] = *(const bf16x8*)(bf1 + k0);
            Bf1.h[1] = *(const bf16x8*)(bf1 + k0 + 8);
            ai0 = wmma_bf16(A.v, Bi0.v, ai0);
            ai1 = wmma_bf16(A.v, Bi1.v, ai1);
            af0 = wmma_bf16(A.v, Bf0.v, af0);
            af1 = wmma_bf16(A.v, Bf1.v, af1);
        }

        // fused gate epilogue for both N tiles
        const size_t tb = (size_t)t * BATCH * NHID;
        #pragma unroll
        for (int e = 0; e < 8; ++e) {
            int    row  = m_base + lh * 8 + e;
            size_t idx0 = (size_t)row * NHID + n_base + lm;
            size_t idx1 = idx0 + 16;

            float iv0 = fast_sigmoid(ai0[e] + ixb[tb + idx0]);
            float fv0 = fast_sigmoid(af0[e] + fxb[tb + idx0]);
            float hn0 = tanhf(iv0 * content[tb + idx0] + fv0 * hread_f[idx0]);
            output[tb + idx0] = hn0;
            hwrite_f[idx0]    = hn0;
            hwrite_bf[idx0]   = f2bf(hn0);

            float iv1 = fast_sigmoid(ai1[e] + ixb[tb + idx1]);
            float fv1 = fast_sigmoid(af1[e] + fxb[tb + idx1]);
            float hn1 = tanhf(iv1 * content[tb + idx1] + fv1 * hread_f[idx1]);
            output[tb + idx1] = hn1;
            hwrite_f[idx1]    = hn1;
            hwrite_bf[idx1]   = f2bf(hn1);

            if (t == T_STEPS - 1) { nexth[idx0] = hn0; nexth[idx1] = hn1; }
        }

        // ---- device-wide barrier between steps ----
        __threadfence();
        __syncthreads();
        if (threadIdx.x == 0) {
            __hip_atomic_fetch_add(counter, 1u, __ATOMIC_RELEASE,
                                   __HIP_MEMORY_SCOPE_AGENT);
            const unsigned target = (unsigned)(t + 1) * SCAN_BLOCKS;
            while (__hip_atomic_load(counter, __ATOMIC_ACQUIRE,
                                     __HIP_MEMORY_SCOPE_AGENT) < target) {
                __builtin_amdgcn_s_sleep(1);
            }
        }
        __syncthreads();
    }
}

// ---------------- host launcher ----------------
extern "C" void kernel_launch(void* const* d_in, const int* in_sizes, int n_in,
                              void* d_out, int out_size, void* d_ws, size_t ws_size,
                              hipStream_t stream) {
    const float* x      = (const float*)d_in[0];
    const float* hidden = (const float*)d_in[1];
    const float* Wcx    = (const float*)d_in[2];
    const float* Wih    = (const float*)d_in[3];
    const float* Wix    = (const float*)d_in[4];
    const float* Wfh    = (const float*)d_in[5];
    const float* Wfx    = (const float*)d_in[6];
    const float* b_i    = (const float*)d_in[7];
    const float* b_f    = (const float*)d_in[8];

    float* output = (float*)d_out;                               // [T,B,H]
    float* nexth  = output + (size_t)T_STEPS * BATCH * NHID;     // [B,H]

    char* ws = (char*)d_ws;
    unsigned short* Wbf     = (unsigned short*)(ws + WBF_OFF);
    unsigned short* xbf     = (unsigned short*)(ws + XBF_OFF);
    unsigned short* hbf     = (unsigned short*)(ws + HBF_OFF);
    float*          hf      = (float*)(ws + HF_OFF);
    unsigned int*   counter = (unsigned int*)(ws + CNT_OFF);
    float*          content = (float*)(ws + CONTENT_OFF);
    float*          ixbuf   = (float*)(ws + IX_OFF);
    float*          fxbuf   = (float*)(ws + FX_OFF);

    // 0) zero barrier counter (deterministic per call)
    init_counter_kernel<<<1, 32, 0, stream>>>(counter);

    // 1) fp32 -> bf16 conversion of x, weights, hidden (one streaming pass)
    convert_kernel<<<8192, 256, 0, stream>>>(x, Wcx, Wih, Wix, Wfh, Wfx, hidden,
                                             xbf, Wbf, hbf, hf);

    // 2) three projection GEMMs: [65536,2048] x [2048,2048]^T
    dim3 pgrid(NHID / 64, (T_STEPS * BATCH) / 128, 3);
    proj_gemm_kernel<<<pgrid, 256, 0, stream>>>(xbf, Wbf, b_i, b_f,
                                                content, ixbuf, fxbuf);

    // 3) persistent recurrent scan (grid barrier per step)
    scan_kernel<<<SCAN_BLOCKS, 256, 0, stream>>>(Wbf, content, ixbuf, fxbuf,
                                                 hbf, hf, output, nexth, counter);
}
